// GrassmannNN_65919158059260
// MI455X (gfx1250) — matrix operand order repaired and checked by reading, not compiled, not run on hardware
//
#include <hip/hip_runtime.h>
#include <hip/hip_bf16.h>

typedef __attribute__((ext_vector_type(16))) _Float16 v16h;
typedef __attribute__((ext_vector_type(8)))  float    v8f;

#define SITE 8

// Native CDNA5 tanh (single TRANS32 op) when available; avoids the branchy
// ocml expansion that bloats VALU count and toggles EXEC between WMMAs.
__device__ __forceinline__ float fast_tanh(float x) {
#if __has_builtin(__builtin_amdgcn_tanhf)
  return __builtin_amdgcn_tanhf(x);
#else
  return tanhf(x);
#endif
}

// ---------------------------------------------------------------------------
// Phase 1 (1 block): fold all Z2-graded masks/signs/permutations + embeddings
// into 28 = 7 sites x 4 (p_old,v) classes of 16x16 matrices W^T, stored
// directly as v_wmma_f32_16x16x32_f16 A operands in this code's contiguous k order:
//   lane L<16: M = L, elem e -> K = e (e = 0..15); lanes 16..31 (K>=16) = 0.
// Also the two possible post-head init states (f16): the head collapses to
// tanh(emb0[v] . head_w[16v: , 16v:]).
// ws halves layout: [0,32) init[2][16] ; [32, 32+28*32*16) A-pack.
// ---------------------------------------------------------------------------
__global__ void grassmann_precompute(const float* __restrict__ emb,     // (8,2,16)
                                     const float* __restrict__ head_w,  // (32,32)
                                     const float* __restrict__ body_w,  // (7,32,32,32)
                                     _Float16* __restrict__ ws_h) {
  const int tid = threadIdx.x;
  // zero the A-pack region (covers the padded K>=16 halves)
  for (int k = tid; k < 28 * 32 * 16; k += blockDim.x) ws_h[32 + k] = (_Float16)0.f;
  // init states
  if (tid < 32) {
    const int v = tid >> 4, t = tid & 15;
    float s = 0.f;
    for (int r = 0; r < 16; ++r)
      s += emb[v * 16 + r] * head_w[(16 * v + r) * 32 + 16 * v + t];
    ws_h[v * 16 + t] = (_Float16)fast_tanh(s);
  }
  __syncthreads();
  // W^T[t][r] = (-1)^(p&v) * sum_u emb_i[v][u] * body_w[i-1][16p+r][16v+u][16(p^v)+t]
  for (int j = tid; j < 28 * 32 * 8; j += blockDim.x) {
    const int m    = j >> 8;          // (site-1)*4 + class
    const int rem  = j & 255;
    const int lane = rem >> 3, e = rem & 7;
    const int site = (m >> 2) + 1, c = m & 3;
    const int p = c >> 1, v = c & 1;
    const int r = 8 * (lane >> 4) + e;   // K index of A operand
    const int t = lane & 15;             // M index of A operand
    const float sgn = (p & v) ? -1.f : 1.f;
    const float* g = body_w + (size_t)(site - 1) * 32 * 32 * 32;
    float s = 0.f;
    for (int u = 0; u < 16; ++u)
      s += emb[(site * 2 + v) * 16 + u] *
           g[((16 * p + r) * 32 + (16 * v + u)) * 32 + (16 * (p ^ v) + t)];
    ws_h[32 + (m * 32 + t) * 16 + 8 * (lane >> 4) + e] = (_Float16)(sgn * s);
  }
}

// ---------------------------------------------------------------------------
// Phase 2: one wave handles 16 batch rows. Per site, 4 class-masked WMMAs:
//   D(16x16, f32) = sum_c  W_c^T (A op)  x  (state ⊙ [class==c]) (B op)
// B operand: lanes 0-15 carry K=0..15 (N = lane%16 = batch column),
// lanes 16-31 (K=16..31) are zero.  D lane L, vgpr d -> t = d+8*(L/16),
// batch = L%16 -> next-step B rebuilt with 4 shfl_xor(16) ops.
// ---------------------------------------------------------------------------
__global__ void grassmann_chain(const int* __restrict__ data,       // (B,8) int32
                                const _Float16* __restrict__ ws_h,
                                float* __restrict__ out) {          // (B,2,32) f32
  const int lane = threadIdx.x & 31;
  const int wave = threadIdx.x >> 5;
  const int row  = lane & 15;
  const int hi   = lane >> 4;
  const int batch = (blockIdx.x * (blockDim.x >> 5) + wave) * 16 + row;

  int bits[SITE];
#pragma unroll
  for (int i = 0; i < SITE; ++i) bits[i] = data[batch * SITE + i];

  const v16h* initPack = (const v16h*)ws_h;        // [2]
  const v16h* aPack    = (const v16h*)(ws_h + 32); // [28*32]

  const _Float16 laneGate = (_Float16)(hi ? 0.f : 1.f);
  v16h b = initPack[bits[0]] * laneGate;           // initial state (post-head tanh)
  int  p = bits[0];                                // active parity sector
  float st[8];

#pragma unroll
  for (int i = 1; i < SITE; ++i) {
    const int v   = bits[i];
    const int myc = p * 2 + v;                     // this row's (p_old,v) class
    v8f acc = {};
#pragma unroll
    for (int c = 0; c < 4; ++c) {
      const v16h a  = aPack[((i - 1) * 4 + c) * 32 + lane];     // shared W_c^T
      const v16h bc = b * (_Float16)((myc == c) ? 1.f : 0.f);   // per-column mask
      acc = __builtin_amdgcn_wmma_f32_16x16x32_f16(false, a, false, bc,
                                                   (short)0, acc, false, false);
    }
    // tanh, then rebuild B operand: lane L<16 needs t=0..7 (own) and t=8..15
    // (from lane L+16); K>=16 lanes zeroed via laneGate.
    union { _Float16 h[8]; int i32[4]; } own, oth;
#pragma unroll
    for (int d = 0; d < 8; ++d) { st[d] = fast_tanh(acc[d]); own.h[d] = (_Float16)st[d]; }
#pragma unroll
    for (int d = 0; d < 4; ++d) oth.i32[d] = __shfl_xor(own.i32[d], 16, 32);
    v16h nb = {};
#pragma unroll
    for (int d = 0; d < 8; ++d) { nb[d] = own.h[d]; nb[8 + d] = oth.h[d]; }
    b = nb * laneGate;
    p ^= v;
  }

  // Output (B,2,32): nonzero only at [p_final][16*p_final + t]; rest must be 0.
  // Lane holds t = 8*hi..8*hi+7 contiguously; lane pair (row,row+16) covers
  // the full 64-float row (16 data + 48 zeros).
  float4* o = (float4*)(out + (size_t)batch * 64);
  const float4 z = make_float4(0.f, 0.f, 0.f, 0.f);
  const int dataOff = (p ? 48 : 0) + 8 * hi;                 // floats
  o[dataOff / 4]     = make_float4(st[0], st[1], st[2], st[3]);
  o[dataOff / 4 + 1] = make_float4(st[4], st[5], st[6], st[7]);
  const int zOff = p ? (24 * hi) : (16 + 24 * hi);           // 24 zero floats/lane
#pragma unroll
  for (int k = 0; k < 6; ++k) o[zOff / 4 + k] = z;
}

extern "C" void kernel_launch(void* const* d_in, const int* in_sizes, int n_in,
                              void* d_out, int out_size, void* d_ws, size_t ws_size,
                              hipStream_t stream) {
  const int*   data  = (const int*)d_in[0];
  const float* emb   = (const float*)d_in[1];
  const float* headw = (const float*)d_in[2];
  const float* bodyw = (const float*)d_in[3];
  _Float16* ws_h = (_Float16*)d_ws;   // needs 28736 bytes
  float* out = (float*)d_out;

  grassmann_precompute<<<1, 256, 0, stream>>>(emb, headw, bodyw, ws_h);
  // 8192 batches / 16 per wave = 512 waves; 4 waves per block -> 128 blocks
  grassmann_chain<<<128, 128, 0, stream>>>(data, ws_h, out);
}
